// EncoderLayer_80135499809170
// MI455X (gfx1250) — matrix operand order repaired and checked by reading, not compiled, not run on hardware
//
#include <hip/hip_runtime.h>

typedef __attribute__((ext_vector_type(16))) _Float16 v16h;
typedef __attribute__((ext_vector_type(2)))  _Float16 h2;
typedef __attribute__((ext_vector_type(8)))  float    v8f;
typedef __attribute__((ext_vector_type(4)))  int      v4i;

#define S_LEN   4096
#define D_MODEL 1024
#define HD      64
#define SCALE   0.125f   // 1/sqrt(64)

#if __has_builtin(__builtin_amdgcn_global_load_async_to_lds_b128)
#define HAVE_ASYNC 1
#else
#define HAVE_ASYNC 0
#endif

typedef __attribute__((address_space(1))) v4i gv4i;   // global int4-vector
typedef __attribute__((address_space(3))) v4i lv4i;   // LDS int4-vector

// one 16-byte chunk: global -> LDS (async if available, else sync copy)
__device__ __forceinline__ void copy16(const void* g, void* l) {
#if HAVE_ASYNC
    __builtin_amdgcn_global_load_async_to_lds_b128(
        (gv4i*)(uintptr_t)g,
        (lv4i*)(unsigned int)(uintptr_t)l, 0, 0);
#else
    *(uint4*)l = *(const uint4*)g;
#endif
}

template <int N>
__device__ __forceinline__ void wait_async() {
#if HAVE_ASYNC
#if __has_builtin(__builtin_amdgcn_s_wait_asynccnt)
    __builtin_amdgcn_s_wait_asynccnt(N);
#else
    asm volatile("s_wait_asynccnt %0" ::"i"(N) : "memory");
#endif
    asm volatile("" ::: "memory");
#endif
}

// A-fragment K index for 16-bit 16x32 A layout (contiguous order):
// lane<16: VGPR0-7 -> K=0..15 ; lane>=16: K=16..31
__device__ __forceinline__ int a_kidx(int j, int half) {
    return half * 16 + 2 * j;
}

// ---------------------------------------------------------------------------
// Kernel 0: pre-swizzle wq (fp32 [64][1024]) into f16 WMMA B-fragments.
// Fragment fid = step*4 + nt (step = k/32, nt = n-tile); per lane 16 contiguous
// k-values of wq row n: wq16[(fid*32 + lane)*16 .. +15]
// ---------------------------------------------------------------------------
__global__ __launch_bounds__(256) void prep_wq_kernel(const float* __restrict__ wq,
                                                      _Float16* __restrict__ wq16) {
    int t    = blockIdx.x * 256 + threadIdx.x;   // 0..4095
    int lane = t & 31;
    int fid  = t >> 5;                           // 0..127
    int step = fid >> 2;
    int nt   = fid & 3;
    int lo   = lane & 15;
    int hi   = lane >> 4;
    int n    = nt * 16 + lo;
    int kb   = step * 32 + hi * 16;
    const float* src = wq + (size_t)n * D_MODEL + kb;
    _Float16 tmp[16];
    #pragma unroll
    for (int j = 0; j < 16; ++j) tmp[j] = (_Float16)src[j];
    uint4* dst = (uint4*)(wq16 + ((size_t)fid * 32 + lane) * 16);
    dst[0] = ((const uint4*)tmp)[0];
    dst[1] = ((const uint4*)tmp)[1];
}

// ---------------------------------------------------------------------------
// Kernel 1: q = x @ wq^T  (f16 WMMA, fp32 accum). 256 WGs x 128 threads;
// wave owns 16 rows x 64 outputs, wave-private async double-buffered x staging.
// Writes q16 [b][s][h] and transposed qT16 [b][h][s].
// ---------------------------------------------------------------------------
#define XS_STRIDE 36   // floats; 144B rows: 16B aligned, bank-spread

__global__ __launch_bounds__(128) void proj_kernel(const float* __restrict__ x,
                                                   const _Float16* __restrict__ wq16,
                                                   _Float16* __restrict__ q16,
                                                   _Float16* __restrict__ qT16) {
    __shared__ float xs[4][2][16 * XS_STRIDE];
    const int wave = threadIdx.x >> 5;
    const int lane = threadIdx.x & 31;
    const int half = lane >> 4;
    const int lo   = lane & 15;
    const int rowbase = blockIdx.x * 64 + wave * 16;   // global b*S+s row

    const float* xbase = x + (size_t)rowbase * D_MODEL;

    auto stage_x = [&](int buf, int ks) {
        #pragma unroll
        for (int i = 0; i < 4; ++i) {
            int idx = lane + 32 * i;          // 0..127 16B chunks
            int r   = idx >> 3;               // row 0..15
            int c   = (idx & 7) * 4;          // col 0..28
            copy16(xbase + (size_t)r * D_MODEL + ks + c,
                   &xs[wave][buf][r * XS_STRIDE + c]);
        }
    };

    v8f acc[4];
    #pragma unroll
    for (int nt = 0; nt < 4; ++nt) acc[nt] = (v8f){};

    stage_x(0, 0);
    stage_x(1, 32);

    for (int ks = 0; ks < D_MODEL; ks += 32) {
        const int buf  = (ks >> 5) & 1;
        const int step = ks >> 5;
        if (ks + 32 < D_MODEL) wait_async<4>(); else wait_async<0>();
        const float* xw = xs[wave][buf];

        // preload all 4 pre-swizzled B fragments (one clause of b128 loads)
        v16h bf[4];
        #pragma unroll
        for (int nt = 0; nt < 4; ++nt)
            bf[nt] = *(const v16h*)(wq16 + ((size_t)(step * 4 + nt) * 32 + lane) * 16);

        // A fragment from LDS (fp32 -> f16)
        v16h a;
        #pragma unroll
        for (int j = 0; j < 8; ++j) {
            int k = a_kidx(j, half);
            a[2 * j]     = (_Float16)xw[lo * XS_STRIDE + k];
            a[2 * j + 1] = (_Float16)xw[lo * XS_STRIDE + k + 1];
        }

        #pragma unroll
        for (int nt = 0; nt < 4; ++nt)
            acc[nt] = __builtin_amdgcn_wmma_f32_16x16x32_f16(
                false, a, false, bf[nt], (short)0, acc[nt], false, false);

        if (ks + 64 < D_MODEL) stage_x(buf, ks + 64);
    }

    const int b  = rowbase >> 12;          // /S_LEN
    const int s0 = rowbase & (S_LEN - 1);
    // row-major q16 [b][s][h]
    #pragma unroll
    for (int nt = 0; nt < 4; ++nt)
        #pragma unroll
        for (int r = 0; r < 8; ++r) {
            int m = r + 8 * half;
            q16[(size_t)(rowbase + m) * HD + nt * 16 + lo] = (_Float16)acc[nt][r];
        }
    // transposed qT16 [b][h][s]: 8 consecutive keys per lane -> one 16B store
    #pragma unroll
    for (int nt = 0; nt < 4; ++nt) {
        _Float16 tmp[8];
        #pragma unroll
        for (int r = 0; r < 8; ++r) tmp[r] = (_Float16)acc[nt][r];
        int h = nt * 16 + lo;
        *(uint4*)(qT16 + ((size_t)b * HD + h) * S_LEN + s0 + 8 * half) =
            *(const uint4*)tmp;
    }
}

// ---------------------------------------------------------------------------
// Kernel 2: flash attention, K=V=Q, online softmax, 64-key tiles.
// 256 WGs x 128 threads (4 waves x 16 queries); double-buffered async staging
// of both row-major (kvK, for scores) and transposed (kvV, for P@V) tiles.
// ---------------------------------------------------------------------------
#define KV_STRIDE 72   // halfs; 144B rows: 16B aligned, bank-spread
#define PB_STRIDE 66

__global__ __launch_bounds__(128) void attn_kernel(const _Float16* __restrict__ q16,
                                                   const _Float16* __restrict__ qT16,
                                                   float* __restrict__ out) {
    __shared__ _Float16 kvK[2][64 * KV_STRIDE];   // [key][dim]
    __shared__ _Float16 kvV[2][64 * KV_STRIDE];   // [dim][key]
    __shared__ _Float16 pb[4][16 * PB_STRIDE];    // per-wave P staging

    const int wave = threadIdx.x >> 5;
    const int lane = threadIdx.x & 31;
    const int half = lane >> 4;
    const int lo   = lane & 15;
    const int wgrow   = blockIdx.x * 64;          // WG base global row
    const int b       = wgrow >> 12;
    const int rowbase = wgrow + wave * 16;        // this wave's first query row
    const _Float16* qb  = q16  + (size_t)b * S_LEN * HD;   // keys, row-major
    const _Float16* qTb = qT16 + (size_t)b * HD * S_LEN;   // keys, [dim][key]

    // stage 64 keys x 64 dims (both layouts): 128 threads x 8 chunks of 16B
    auto stage_kv = [&](int buf, int kt) {
        #pragma unroll
        for (int i = 0; i < 4; ++i) {
            int idx = threadIdx.x * 4 + i;        // 0..511 chunks
            int r   = idx >> 3;                   // key 0..63
            int c   = (idx & 7) * 8;              // dim 0..56
            copy16(qb + (size_t)(kt + r) * HD + c,
                   &kvK[buf][r * KV_STRIDE + c]);
        }
        #pragma unroll
        for (int i = 0; i < 4; ++i) {
            int idx = threadIdx.x * 4 + i;
            int d   = idx >> 3;                   // dim 0..63
            int c   = (idx & 7) * 8;              // key offset 0..56
            copy16(qTb + (size_t)d * S_LEN + kt + c,
                   &kvV[buf][d * KV_STRIDE + c]);
        }
    };

    // Q A-fragments (head dim 0-31, 32-63), pre-scaled by 1/sqrt(64)
    v16h qa[2];
    {
        const _Float16* qrow = q16 + (size_t)(rowbase + lo) * HD;
        #pragma unroll
        for (int f = 0; f < 2; ++f)
            #pragma unroll
            for (int j = 0; j < 8; ++j) {
                int k = f * 32 + a_kidx(j, half);
                qa[f][2 * j]     = (_Float16)(SCALE * (float)qrow[k]);
                qa[f][2 * j + 1] = (_Float16)(SCALE * (float)qrow[k + 1]);
            }
    }

    float mrun[8], lrun[8];
    #pragma unroll
    for (int r = 0; r < 8; ++r) { mrun[r] = -3.0e38f; lrun[r] = 0.0f; }
    v8f o[4];
    #pragma unroll
    for (int nt = 0; nt < 4; ++nt) o[nt] = (v8f){};

    stage_kv(0, 0);
    stage_kv(1, 64);

    for (int kt = 0; kt < S_LEN; kt += 64) {
        const int buf = (kt >> 6) & 1;
        if (kt + 64 < S_LEN) wait_async<8>(); else wait_async<0>();
        __syncthreads();                           // whole tile resident
        const _Float16* kk = kvK[buf];
        const _Float16* vv = kvV[buf];

        // ---- scores: four 16x16 tiles over this 64-key block ----
        v8f sc[4];
        #pragma unroll
        for (int st = 0; st < 4; ++st) {
            sc[st] = (v8f){};
            int n = st * 16 + lo;                 // key column
            #pragma unroll
            for (int f = 0; f < 2; ++f) {
                int kb = f * 32 + half * 16;      // contraction (head-dim) base
                v16h bf;
                #pragma unroll
                for (int j = 0; j < 8; ++j) {
                    h2 w = *(const h2*)&kk[n * KV_STRIDE + kb + 2 * j];
                    bf[2 * j]     = w.x;
                    bf[2 * j + 1] = w.y;
                }
                sc[st] = __builtin_amdgcn_wmma_f32_16x16x32_f16(
                    false, qa[f], false, bf, (short)0, sc[st], false, false);
            }
        }

        // ---- online softmax (row = r + 8*half, cols across 16 lanes) ----
        float alpha[8];
        #pragma unroll
        for (int r = 0; r < 8; ++r) {
            float t = fmaxf(fmaxf(sc[0][r], sc[1][r]), fmaxf(sc[2][r], sc[3][r]));
            #pragma unroll
            for (int off = 1; off < 16; off <<= 1)
                t = fmaxf(t, __shfl_xor(t, off, 32));
            float newm = fmaxf(mrun[r], t);
            alpha[r] = __expf(mrun[r] - newm);
            mrun[r]  = newm;
            float rs = 0.0f;
            #pragma unroll
            for (int st = 0; st < 4; ++st) {
                float p = __expf(sc[st][r] - newm);
                sc[st][r] = p;
                rs += p;
            }
            #pragma unroll
            for (int off = 1; off < 16; off <<= 1)
                rs += __shfl_xor(rs, off, 32);
            lrun[r] = lrun[r] * alpha[r] + rs;
        }
        #pragma unroll
        for (int nt = 0; nt < 4; ++nt)
            #pragma unroll
            for (int r = 0; r < 8; ++r)
                o[nt][r] *= alpha[r];

        // ---- P: C layout -> LDS -> A layout (per-wave region) ----
        _Float16* pw = pb[wave];
        #pragma unroll
        for (int st = 0; st < 4; ++st)
            #pragma unroll
            for (int r = 0; r < 8; ++r) {
                int m = r + 8 * half;
                pw[m * PB_STRIDE + st * 16 + lo] = (_Float16)sc[st][r];
            }
        v16h pa[2];
        #pragma unroll
        for (int f = 0; f < 2; ++f)
            #pragma unroll
            for (int j = 0; j < 8; ++j) {
                int k = f * 32 + a_kidx(j, half);
                h2 w = *(const h2*)&pw[lo * PB_STRIDE + k];
                pa[f][2 * j]     = w.x;
                pa[f][2 * j + 1] = w.y;
            }

        // ---- O += P @ V  (V fragments from transposed tile: paired loads) --
        #pragma unroll
        for (int nt = 0; nt < 4; ++nt) {
            int n = nt * 16 + lo;                 // head-dim column
            #pragma unroll
            for (int f = 0; f < 2; ++f) {
                int kb = f * 32 + half * 16;      // key contraction base
                v16h vb;
                #pragma unroll
                for (int j = 0; j < 8; ++j) {
                    h2 w = *(const h2*)&vv[n * KV_STRIDE + kb + 2 * j];
                    vb[2 * j]     = w.x;
                    vb[2 * j + 1] = w.y;
                }
                o[nt] = __builtin_amdgcn_wmma_f32_16x16x32_f16(
                    false, pa[f], false, vb, (short)0, o[nt], false, false);
            }
        }

        __syncthreads();                           // all waves done with buf
        if (kt + 128 < S_LEN) stage_kv(buf, kt + 128);
    }

    // ---- normalize and store fp32 output ----
    #pragma unroll
    for (int r = 0; r < 8; ++r) {
        int m = r + 8 * half;
        float inv = 1.0f / lrun[r];
        #pragma unroll
        for (int nt = 0; nt < 4; ++nt)
            out[(size_t)(rowbase + m) * HD + nt * 16 + lo] = o[nt][r] * inv;
    }
}

// ---------------------------------------------------------------------------
extern "C" void kernel_launch(void* const* d_in, const int* in_sizes, int n_in,
                              void* d_out, int out_size, void* d_ws, size_t ws_size,
                              hipStream_t stream) {
    const float* x  = (const float*)d_in[0];   // [4,4096,1024] fp32
    const float* wq = (const float*)d_in[1];   // [64,1024] fp32

    char* ws = (char*)d_ws;
    _Float16* q16  = (_Float16*)ws;                         // 2 MB
    _Float16* qT16 = (_Float16*)(ws + (2u << 20));          // 2 MB
    _Float16* wq16 = (_Float16*)(ws + (4u << 20));          // 128 KB
    float* out = (float*)d_out;                             // [4,4096,64] fp32

    (void)in_sizes; (void)n_in; (void)out_size; (void)ws_size;

    prep_wq_kernel<<<16, 256, 0, stream>>>(wq, wq16);
    proj_kernel<<<256, 128, 0, stream>>>(x, wq16, q16, qT16);
    attn_kernel<<<256, 128, 0, stream>>>(q16, qT16, out);
}
